// LlamaAttention_71648644432090
// MI455X (gfx1250) — compile-verified, matrix operand order checked by reading, not run on hardware
//
#include <hip/hip_runtime.h>
#include <hip/hip_bf16.h>

// ---- WMMA vector types (CDNA5 / gfx1250, wave32) ----
typedef __attribute__((ext_vector_type(8)))  __bf16 v8bf;
typedef __attribute__((ext_vector_type(16))) __bf16 v16bf;
typedef __attribute__((ext_vector_type(8)))  float  v8f;

__device__ __forceinline__ __bf16 f2bf(float f) {
    unsigned u = __builtin_bit_cast(unsigned, f);
    u += 0x7FFFu + ((u >> 16) & 1u);           // round-to-nearest-even
    unsigned short h = (unsigned short)(u >> 16);
    return __builtin_bit_cast(__bf16, h);
}

// round-to-nearest-even pack of two fp32 into one dword of two bf16 (lo, hi)
__device__ __forceinline__ unsigned pack2bf(float lo, float hi) {
    unsigned ul = __builtin_bit_cast(unsigned, lo);
    unsigned uh = __builtin_bit_cast(unsigned, hi);
    ul += 0x7FFFu + ((ul >> 16) & 1u);
    uh += 0x7FFFu + ((uh >> 16) & 1u);
    return (ul >> 16) | (uh & 0xFFFF0000u);
}

// Fragment loader matching the documented 16-bit A/B layout:
// lane m (0..15)/m+16 holds row M=m; half = lane>>4 selects K sub-range.
// Elements e<8 -> K = base + half*8 + e ; e>=8 -> K = base + 16 + half*8 + (e-8).
// "row" must point at the K-window base (16B aligned), off = half*8.
__device__ __forceinline__ v16bf load_frag(const __bf16* row, int off) {
    v8bf lo = *(const v8bf*)(row + off);
    v8bf hi = *(const v8bf*)(row + off + 16);
    return __builtin_shufflevector(lo, hi, 0,1,2,3,4,5,6,7,8,9,10,11,12,13,14,15);
}

// ================== GEMM: C[M,N] = A[M,K] * W[K,N], fp32 io, bf16 WMMA ==================
// block = 128 threads (4 waves), tile 64x64, wave w owns N-cols [16w,16w+16), 4 M-tiles.
// Register double-buffered; LDS commits are packed dword stores (2 bf16 per ds_store_b32).
__global__ __launch_bounds__(128) void gemm_bf16w(const float* __restrict__ A,
                                                  const float* __restrict__ W,
                                                  float* __restrict__ C,
                                                  int M, int K, int N) {
    __shared__ __align__(16) __bf16 As[64][40];   // [m][k], rows padded (80B = 5*16B)
    __shared__ __align__(16) __bf16 Ws[64][40];   // [n][k] (transposed on load)
    const int tid  = threadIdx.x;
    const int lane = tid & 31, w = tid >> 5;
    const int half = lane >> 4, lm = lane & 15;
    const int m0 = blockIdx.x * 64, n0 = blockIdx.y * 64;

    // per-thread PAIR coordinates (fixed across K-steps); 1024 pairs = 8 iters x 128 thr
    int ar[8], ac[8];   // A: row, K-pair base (pair along K, row-major LDS)
    int wn[8], wk[8];   // W: n, K-pair base (pair along K of transposed LDS tile)
    #pragma unroll
    for (int i = 0; i < 8; ++i) {
        int p = i * 128 + tid;
        ar[i] = p >> 4;        ac[i] = (p & 15) * 2;
        wn[i] = p & 63;        wk[i] = (p >> 6) * 2;
    }

    float2 areg[8];
    float  w0reg[8], w1reg[8];
    #pragma unroll
    for (int i = 0; i < 8; ++i) {                 // prologue: fetch tile kb=0
        areg[i]  = *(const float2*)&A[(size_t)(m0 + ar[i]) * K + ac[i]];
        w0reg[i] = W[(size_t)(wk[i] + 0) * N + n0 + wn[i]];
        w1reg[i] = W[(size_t)(wk[i] + 1) * N + n0 + wn[i]];
    }

    v8f acc[4] = {};
    for (int kb = 0; kb < K; kb += 32) {
        #pragma unroll
        for (int i = 0; i < 8; ++i) {             // commit current tile to LDS (packed bf16)
            *(unsigned*)&As[ar[i]][ac[i]] = pack2bf(areg[i].x, areg[i].y);
            *(unsigned*)&Ws[wn[i]][wk[i]] = pack2bf(w0reg[i], w1reg[i]);
        }
        __syncthreads();
        if (kb + 32 < K) {                        // prefetch next tile -> VGPRs
            #pragma unroll                        // (overlaps the WMMAs below)
            for (int i = 0; i < 8; ++i) {
                areg[i]  = *(const float2*)&A[(size_t)(m0 + ar[i]) * K + (kb + 32) + ac[i]];
                w0reg[i] = W[(size_t)(kb + 32 + wk[i] + 0) * N + n0 + wn[i]];
                w1reg[i] = W[(size_t)(kb + 32 + wk[i] + 1) * N + n0 + wn[i]];
            }
        }
        v16bf bfrag = load_frag(&Ws[16 * w + lm][0], half * 8);
        #pragma unroll
        for (int mt = 0; mt < 4; ++mt) {
            v16bf afrag = load_frag(&As[16 * mt + lm][0], half * 8);
            acc[mt] = __builtin_amdgcn_wmma_f32_16x16x32_bf16(
                false, afrag, false, bfrag, (short)0, acc[mt], false, false);
        }
        __syncthreads();
    }
    #pragma unroll
    for (int mt = 0; mt < 4; ++mt)
        #pragma unroll
        for (int r = 0; r < 8; ++r)
            C[(size_t)(m0 + 16 * mt + r + 8 * half) * N + n0 + 16 * w + lm] = acc[mt][r];
}

// ================== RoPE: in-place, each thread owns a (d, d+64) pair ==================
__global__ __launch_bounds__(256) void rope_kernel(float* __restrict__ x,
                                                   const int* __restrict__ pos_ids,
                                                   int T, int H) {
    int idx = blockIdx.x * blockDim.x + threadIdx.x;     // over T*H*64
    int d  = idx & 63;
    int th = idx >> 6;
    int hh = th % H, t = th / H;
    if (t >= T) return;
    float* p = x + ((size_t)t * H + hh) * 128;
    float pos = (float)pos_ids[t];
    // inv_freq = theta^(-d/64); ln(10000)/64
    float ang = pos * __expf(-(float)d * (9.2103403719761836f / 64.0f));
    float cs = __cosf(ang), sn = __sinf(ang);
    float x0 = p[d], x1 = p[d + 64];
    p[d]      = x0 * cs - x1 * sn;
    p[d + 64] = x1 * cs + x0 * sn;
}

// ================== Paged flash attention: 1 block per (head, seq) ==================
// 4 waves; wave w owns query rows [16w,16w+16). 33 chunks of 64 keys (chunk==cache block);
// chunk 32 = new tokens (from workspace, causal-masked).
__global__ __launch_bounds__(128) void attn_kernel(const float* __restrict__ q,
                                                   const float* __restrict__ knew,
                                                   const float* __restrict__ vnew,
                                                   const float* __restrict__ kc,
                                                   const float* __restrict__ vc,
                                                   const int* __restrict__ blkoff,
                                                   float* __restrict__ out) {
    __shared__ __align__(16) __bf16 Qs[64][136];                 // [query][d]
    __shared__ __align__(16) unsigned char KSU[64 * 136 * 2];    // Ks [64][136] bf16  UNION  Ss [64][68] f32
    __shared__ __align__(16) __bf16 Vts[128][72];                // [d][key]  (transposed)
    __shared__ __align__(16) __bf16 Ps[64][72];                  // softmaxed probs
    __shared__ float rowM[64], rowL[64], rowAlpha[64];
    __shared__ float pmax[128], psum[128];                       // split-row softmax partials
    __bf16 (*Ks)[136] = (__bf16(*)[136])KSU;
    float  (*Ss)[68]  = (float(*)[68])KSU;

    const int tid  = threadIdx.x, lane = tid & 31, w = tid >> 5;
    const int half = lane >> 4,  lm   = lane & 15;
    const int h = blockIdx.x, b = blockIdx.y;

    for (int i = 0; i < 32; ++i) {               // load Q once (64x128, packed pairs along d)
        int p = i * 128 + tid;
        int r = p >> 6, db = (p & 63) * 2;
        float2 qv = *(const float2*)&q[((size_t)(b * 64 + r) * 32 + h) * 128 + db];
        *(unsigned*)&Qs[r][db] = pack2bf(qv.x, qv.y);
    }
    if (tid < 64) { rowM[tid] = -3.0e38f; rowL[tid] = 0.0f; }
    v8f oacc[8] = {};
    __syncthreads();

    for (int c = 0; c < 33; ++c) {
        const float *kp, *vp;
        if (c < 32) {
            int blk = blkoff[b * 33 + c];
            kp = kc + ((size_t)blk * 64 * 32 + h) * 128;
            vp = vc + ((size_t)blk * 64 * 32 + h) * 128;
        } else {
            kp = knew + ((size_t)(b * 64) * 32 + h) * 128;
            vp = vnew + ((size_t)(b * 64) * 32 + h) * 128;
        }
        if (c + 1 < 33) {                        // prefetch next chunk's K block
            const float* np = (c + 1 < 32)
                ? kc + ((size_t)blkoff[b * 33 + c + 1] * 64 * 32 + h) * 128
                : knew + ((size_t)(b * 64) * 32 + h) * 128;
            __builtin_prefetch(np + tid, 0, 1);
        }
        for (int i = 0; i < 32; ++i) {           // K chunk -> LDS [key][d], packed pairs along d
            int p = i * 128 + tid;
            int key = p >> 6, db = (p & 63) * 2;
            float2 kv = *(const float2*)&kp[(size_t)key * 4096 + db];
            *(unsigned*)&Ks[key][db] = pack2bf(kv.x, kv.y);
        }
        for (int i = 0; i < 32; ++i) {           // V chunk -> LDS transposed [d][key], pairs along key
            int p = i * 128 + tid;
            int d = p & 127, kb2 = (p >> 7) * 2;
            float v0 = vp[(size_t)(kb2 + 0) * 4096 + d];
            float v1 = vp[(size_t)(kb2 + 1) * 4096 + d];
            *(unsigned*)&Vts[d][kb2] = pack2bf(v0, v1);
        }
        __syncthreads();

        // S = Q K^T : rows 16w..16w+15 x 64 keys, K-dim = 128 in 4 steps
        v8f sacc[4] = {};
        #pragma unroll
        for (int ks = 0; ks < 4; ++ks) {
            v16bf a = load_frag(&Qs[16 * w + lm][ks * 32], half * 8);
            #pragma unroll
            for (int nt = 0; nt < 4; ++nt) {
                v16bf bk = load_frag(&Ks[16 * nt + lm][ks * 32], half * 8);
                sacc[nt] = __builtin_amdgcn_wmma_f32_16x16x32_bf16(
                    false, a, false, bk, (short)0, sacc[nt], false, false);
            }
        }
        __syncthreads();                          // all waves done reading Ks (Ss aliases it)
        #pragma unroll
        for (int nt = 0; nt < 4; ++nt)
            #pragma unroll
            for (int r = 0; r < 8; ++r)
                Ss[16 * w + r + 8 * half][16 * nt + lm] = sacc[nt][r] * 0.08838834764831845f;
        __syncthreads();

        // streaming softmax: 2 threads per row, 32 columns each, all 128 threads busy
        {
            const int s = tid >> 1, hh = tid & 1;
            const int c0 = hh * 32;
            const int lim = (c == 32) ? (s + 1) : 64;      // causal mask on new-token chunk
            float lmax = -3.0e38f;
            for (int j = c0; j < c0 + 32; ++j)
                if (j < lim) lmax = fmaxf(lmax, Ss[s][j]);
            pmax[tid] = lmax;
            __syncthreads();
            float m  = rowM[s];
            float nm = fmaxf(m, fmaxf(pmax[2 * s], pmax[2 * s + 1]));
            float lsum = 0.0f;
            for (int j = c0; j < c0 + 32; j += 2) {        // packed dword stores to Ps
                float pv0 = (j + 0 < lim) ? __expf(Ss[s][j + 0] - nm) : 0.0f;
                float pv1 = (j + 1 < lim) ? __expf(Ss[s][j + 1] - nm) : 0.0f;
                lsum += pv0 + pv1;
                *(unsigned*)&Ps[s][j] = pack2bf(pv0, pv1);
            }
            psum[tid] = lsum;
            __syncthreads();
            if (hh == 0) {
                float alpha = __expf(m - nm);
                rowAlpha[s] = alpha;
                rowM[s]     = nm;
                rowL[s]     = rowL[s] * alpha + psum[2 * s] + psum[2 * s + 1];
            }
        }
        __syncthreads();

        // rescale O, then O += P V
        float al[8];
        #pragma unroll
        for (int r = 0; r < 8; ++r) al[r] = rowAlpha[16 * w + r + 8 * half];
        #pragma unroll
        for (int dt = 0; dt < 8; ++dt)
            #pragma unroll
            for (int r = 0; r < 8; ++r) oacc[dt][r] *= al[r];

        v16bf pa[2];
        #pragma unroll
        for (int ks = 0; ks < 2; ++ks) pa[ks] = load_frag(&Ps[16 * w + lm][ks * 32], half * 8);
        #pragma unroll
        for (int dt = 0; dt < 8; ++dt)
            #pragma unroll
            for (int ks = 0; ks < 2; ++ks) {
                v16bf bv = load_frag(&Vts[16 * dt + lm][ks * 32], half * 8);
                oacc[dt] = __builtin_amdgcn_wmma_f32_16x16x32_bf16(
                    false, pa[ks], false, bv, (short)0, oacc[dt], false, false);
            }
        __syncthreads();
    }

    float il[8];
    #pragma unroll
    for (int r = 0; r < 8; ++r) il[r] = 1.0f / rowL[16 * w + r + 8 * half];
    #pragma unroll
    for (int dt = 0; dt < 8; ++dt)
        #pragma unroll
        for (int r = 0; r < 8; ++r) {
            int row = 16 * w + r + 8 * half;
            out[((size_t)(b * 64 + row) * 32 + h) * 128 + 16 * dt + lm] = oacc[dt][r] * il[r];
        }
}

// ================== host-side orchestration ==================
extern "C" void kernel_launch(void* const* d_in, const int* in_sizes, int n_in,
                              void* d_out, int out_size, void* d_ws, size_t ws_size,
                              hipStream_t stream) {
    (void)in_sizes; (void)n_in; (void)out_size; (void)ws_size;
    const float* hidden = (const float*)d_in[0];
    const float* kcache = (const float*)d_in[1];
    const float* vcache = (const float*)d_in[2];
    const float* wq     = (const float*)d_in[3];
    const float* wk     = (const float*)d_in[4];
    const float* wv     = (const float*)d_in[5];
    const float* wo     = (const float*)d_in[6];
    const int*   pos    = (const int*)d_in[7];
    const int*   blkoff = (const int*)d_in[8];
    float* out = (float*)d_out;

    const size_t TN = (size_t)512 * 4096;
    float* q_ws = (float*)d_ws;
    float* k_ws = q_ws + TN;
    float* v_ws = k_ws + TN;
    float* a_ws = v_ws + TN;

    dim3 gg(8, 64), bb(128);
    gemm_bf16w<<<gg, bb, 0, stream>>>(hidden, wq, q_ws, 512, 4096, 4096);
    gemm_bf16w<<<gg, bb, 0, stream>>>(hidden, wk, k_ws, 512, 4096, 4096);
    gemm_bf16w<<<gg, bb, 0, stream>>>(hidden, wv, v_ws, 512, 4096, 4096);
    rope_kernel<<<4096, 256, 0, stream>>>(q_ws, pos, 512, 32);
    rope_kernel<<<4096, 256, 0, stream>>>(k_ws, pos, 512, 32);
    attn_kernel<<<dim3(32, 8), bb, 0, stream>>>(q_ws, k_ws, v_ws, kcache, vcache, blkoff, a_ws);
    gemm_bf16w<<<gg, bb, 0, stream>>>(a_ws, wo, out, 512, 4096, 4096);
}
